// VRNNGATA2C_63299228009035
// MI455X (gfx1250) — matrix operand order repaired and checked by reading, not compiled, not run on hardware
//
#include <hip/hip_runtime.h>
#include <math.h>

// ---------------------------------------------------------------------------
// CDNA5 (gfx1250) VRNN + GAT + per-agent A2C step.
//  * Shared-weight GEMMs: v_wmma_f32_16x16x32_bf16, operands pre-converted to
//    bf16 (weights once at start; chained activations written as bf16 by their
//    producers) -> inner loop is pure b128 loads + WMMA, A-fragment reused
//    across two N-tiles per wave.
//  * Per-agent heads: 268MB unique fp32 weights, pure HBM streaming ->
//    coalesced float4 GEMV, wave32 shuffle reductions, global_prefetch.
//  * Pairwise-KL coop collapsed from O(N^2 Z) to O(N Z) via column sums.
// ---------------------------------------------------------------------------

typedef __attribute__((ext_vector_type(16))) __bf16 v16bf;
typedef __attribute__((ext_vector_type(8)))  __bf16 v8bf;
typedef __attribute__((ext_vector_type(4)))  __bf16 v4bf;
typedef __attribute__((ext_vector_type(8)))  float  v8f;

__device__ __forceinline__ float clamp10(float x) { return fminf(fmaxf(x, -10.f), 10.f); }
__device__ __forceinline__ float sigmoidf(float x) { return 1.f / (1.f + __expf(-x)); }

__device__ __forceinline__ float block_reduce_sum(float v, float* sm) {
  int tid = threadIdx.x;
  sm[tid] = v; __syncthreads();
  for (int s = blockDim.x >> 1; s > 0; s >>= 1) {
    if (tid < s) sm[tid] += sm[tid + s];
    __syncthreads();
  }
  float r = sm[0]; __syncthreads();
  return r;
}

__device__ __forceinline__ float block_reduce_max(float v, float* sm) {
  int tid = threadIdx.x;
  sm[tid] = v; __syncthreads();
  for (int s = blockDim.x >> 1; s > 0; s >>= 1) {
    if (tid < s) sm[tid] = fmaxf(sm[tid], sm[tid + s]);
    __syncthreads();
  }
  float r = sm[0]; __syncthreads();
  return r;
}

// fp32 -> bf16 bulk conversion (float4 / 4-wide)
__global__ void cvt4_bf16_kernel(const float* __restrict__ src, __bf16* __restrict__ dst, int n4) {
  int i = blockIdx.x * blockDim.x + threadIdx.x;
  if (i >= n4) return;
  float4 p = ((const float4*)src)[i];
  v4bf o;
  o[0] = (__bf16)p.x; o[1] = (__bf16)p.y; o[2] = (__bf16)p.z; o[3] = (__bf16)p.w;
  ((v4bf*)dst)[i] = o;
}

// ---------------------------------------------------------------------------
// WMMA bf16 GEMM:  Out(MxN) = act( X(MxK) @ W(NxK)^T + bias )
// bf16 operands in memory. One wave computes two adjacent 16x16 N-tiles
// (A fragment reused); 4 waves per block along N; K unrolled by 64 -> 4 WMMA.
// Fragment layouts used here (contiguous k, same for A and B):
//   A 16x32: row = lane&15, elem e -> K = (lane>>4)*16 + e. (one 32B load)
//   B 32x16: col = lane&15, elem e -> K = (lane>>4)*16 + e. (one 32B load)
//   C/D f32: VGPR r -> row (lane>>4)*8 + r, col = lane&15.
// ---------------------------------------------------------------------------
__global__ __launch_bounds__(128)
void gemm_bf16_wmma(const __bf16* __restrict__ X, const __bf16* __restrict__ W,
                    const float* __restrict__ bias, float* __restrict__ Out,
                    __bf16* __restrict__ OutBf, int Nn, int K, int act) {
  int wave = threadIdx.x >> 5;
  int lane = threadIdx.x & 31;
  int half = lane >> 4;
  int lm   = lane & 15;
  int n0 = (blockIdx.x * 4 + wave) * 32;       // two 16-col tiles per wave
  int m0 = blockIdx.y * 16;

  const __bf16* Arow  = X + (size_t)(m0 + lm) * K;
  const __bf16* Brow0 = W + (size_t)(n0 + lm) * K + half * 16;
  const __bf16* Brow1 = W + (size_t)(n0 + 16 + lm) * K + half * 16;

  v8f acc0 = {}, acc1 = {};
  for (int k0 = 0; k0 < K; k0 += 64) {         // K always a multiple of 64
#pragma unroll
    for (int kk = 0; kk < 2; ++kk) {
      int kb = k0 + kk * 32;
      v16bf a = *(const v16bf*)(Arow + kb + half * 16);
      v16bf b0 = *(const v16bf*)(Brow0 + kb);
      v16bf b1 = *(const v16bf*)(Brow1 + kb);
      acc0 = __builtin_amdgcn_wmma_f32_16x16x32_bf16(false, a, false, b0,
                                                     (short)0, acc0, false, false);
      acc1 = __builtin_amdgcn_wmma_f32_16x16x32_bf16(false, a, false, b1,
                                                     (short)0, acc1, false, false);
    }
  }
  float bv0 = bias ? bias[n0 + lm] : 0.f;
  float bv1 = bias ? bias[n0 + 16 + lm] : 0.f;
#pragma unroll
  for (int r = 0; r < 8; ++r) {
    int m = m0 + half * 8 + r;
    float v0 = acc0[r] + bv0;
    float v1 = acc1[r] + bv1;
    if (act) { v0 = fmaxf(v0, 0.f); v1 = fmaxf(v1, 0.f); }
    size_t o0 = (size_t)m * Nn + n0 + lm;
    size_t o1 = o0 + 16;
    if (Out)   { Out[o0] = v0;            Out[o1] = v1; }
    if (OutBf) { OutBf[o0] = (__bf16)v0;  OutBf[o1] = (__bf16)v1; }
  }
}

// ---------------------------------------------------------------------------
// Elementwise / reduction kernels
// ---------------------------------------------------------------------------
__global__ void concat2_bf16_kernel(__bf16* __restrict__ dst, const float* __restrict__ a, int wa,
                                    const float* __restrict__ b, int wb, int total) {
  int idx = blockIdx.x * blockDim.x + threadIdx.x;
  if (idx >= total) return;
  int w = wa + wb;
  int r = idx / w, c = idx - r * w;
  float v = (c < wa) ? a[(size_t)r * wa + c] : b[(size_t)r * wb + (c - wa)];
  dst[idx] = (__bf16)v;
}

__global__ void concat2_kernel(float* __restrict__ dst, const float* __restrict__ a, int wa,
                               const float* __restrict__ b, int wb, int total) {
  int idx = blockIdx.x * blockDim.x + threadIdx.x;
  if (idx >= total) return;
  int w = wa + wb;
  int r = idx / w, c = idx - r * w;
  dst[idx] = (c < wa) ? a[(size_t)r * wa + c] : b[(size_t)r * wb + (c - wa)];
}

// mu_q / clipped lv_q / z from encoder output (N x 2Z)
__global__ void vrnnz_kernel(const float* __restrict__ eout, const float* __restrict__ eps,
                             float* __restrict__ muq, float* __restrict__ lvq,
                             float* __restrict__ z, int Zd, int total) {
  int idx = blockIdx.x * blockDim.x + threadIdx.x;
  if (idx >= total) return;
  int i = idx / Zd, d = idx - i * Zd;
  float mu = eout[(size_t)i * 2 * Zd + d];
  float lv = clamp10(eout[(size_t)i * 2 * Zd + Zd + d]);
  muq[idx] = mu;
  lvq[idx] = lv;
  z[idx] = mu + eps[idx] * __expf(0.5f * lv);
}

__global__ void nll_kernel(const float* __restrict__ obs, const float* __restrict__ dx,
                           float* __restrict__ nll, int OBSd) {
  __shared__ float sm[256];
  int i = blockIdx.x, t = threadIdx.x;
  float mu = dx[(size_t)i * 2 * OBSd + t];
  float lv = clamp10(dx[(size_t)i * 2 * OBSd + OBSd + t]);
  float df = obs[(size_t)i * OBSd + t] - mu;
  float s = block_reduce_sum(df * df * __expf(-lv) + lv, sm);
  if (t == 0) nll[i] = 0.5f * s;
}

__global__ void kl_kernel(const float* __restrict__ pout, const float* __restrict__ muq,
                          const float* __restrict__ lvq, float* __restrict__ kl, int Zd) {
  __shared__ float sm[128];
  int i = blockIdx.x, t = threadIdx.x;
  float mup = pout[(size_t)i * 2 * Zd + t];
  float lvp = clamp10(pout[(size_t)i * 2 * Zd + Zd + t]);
  float mq = muq[(size_t)i * Zd + t];
  float lq = lvq[(size_t)i * Zd + t];
  float ivp = __expf(-lvp);
  float v = (mq - mup) * (mq - mup) * ivp + __expf(lq) * ivp + (lvp - lq) - 1.f;
  float s = block_reduce_sum(v, sm);
  if (t == 0) kl[i] = 0.5f * s;
}

__global__ void gru_kernel(const float* __restrict__ gi, const float* __restrict__ gh,
                           const float* __restrict__ hp, float* __restrict__ hnew,
                           __bf16* __restrict__ hnew_bf, int H, int total) {
  int idx = blockIdx.x * blockDim.x + threadIdx.x;
  if (idx >= total) return;
  int n = idx / H, h = idx - n * H;
  const float* gin = gi + (size_t)n * 3 * H;
  const float* ghn = gh + (size_t)n * 3 * H;
  float r = sigmoidf(gin[h] + ghn[h]);
  float u = sigmoidf(gin[H + h] + ghn[H + h]);
  float c = tanhf(gin[2 * H + h] + r * ghn[2 * H + h]);
  float hv = hp[idx];
  float v = (1.f - u) * c + u * hv;
  hnew[idx] = v;
  hnew_bf[idx] = (__bf16)v;
}

// column sums for closed-form coop: S1_d = sum_j e^{-lv}, S2_d = sum_j mu e^{-lv},
// scal[0] = sum_{j,d} mu^2 e^{-lv}, scal[1] = sum_{j,d} lv
__global__ void klstats_kernel(const float* __restrict__ muq, const float* __restrict__ lvq,
                               float* __restrict__ S1, float* __restrict__ S2,
                               float* __restrict__ scal, int Nn, int Zd) {
  __shared__ float sm[128];
  int d = threadIdx.x;
  float s1 = 0.f, s2 = 0.f, c3 = 0.f, sl = 0.f;
  for (int i = 0; i < Nn; ++i) {
    float mu = muq[(size_t)i * Zd + d];
    float lv = lvq[(size_t)i * Zd + d];
    float iv = __expf(-lv);
    s1 += iv; s2 += mu * iv; c3 += mu * mu * iv; sl += lv;
  }
  S1[d] = s1; S2[d] = s2;
  float c3t = block_reduce_sum(c3, sm);
  float slt = block_reduce_sum(sl, sm);
  if (d == 0) { scal[0] = c3t; scal[1] = slt; }
}

// coop_i = 0.5/(n-1) * [ sum_d (mu^2+e^lv)*S1 - 2 mu*S2 + C3 + SL - n*slv_i - n*Z ]
__global__ void coop_kernel(const float* __restrict__ muq, const float* __restrict__ lvq,
                            const float* __restrict__ S1, const float* __restrict__ S2,
                            const float* __restrict__ scal, const float* __restrict__ cw,
                            const float* __restrict__ cb, float* __restrict__ cn,
                            int Nn, int Zd) {
  __shared__ float sm[128];
  int i = blockIdx.x, d = threadIdx.x;
  float mu = muq[(size_t)i * Zd + d];
  float lv = lvq[(size_t)i * Zd + d];
  float t = (mu * mu + __expf(lv)) * S1[d] - 2.f * mu * S2[d];
  float red = block_reduce_sum(t, sm);
  float slv = block_reduce_sum(lv, sm);
  if (d == 0) {
    float total = red + scal[0] + scal[1] - (float)Nn * slv - (float)Nn * (float)Zd;
    float coop = 0.5f * total / (float)(Nn - 1);
    cn[i] = sigmoidf(cw[0] * coop + cb[0]);
  }
}

__global__ void gat_scores_kernel(const float* __restrict__ Wh, const float* __restrict__ asrc,
                                  const float* __restrict__ adst, float* __restrict__ ssrc,
                                  float* __restrict__ sdst, int d) {
  __shared__ float sm[256];
  int i = blockIdx.x, t = threadIdx.x;
  float w = Wh[(size_t)i * d + t];
  float a = block_reduce_sum(w * asrc[t], sm);
  float b = block_reduce_sum(w * adst[t], sm);
  if (t == 0) { ssrc[i] = a; sdst[i] = b; }
}

__global__ void gat_agg_kernel(const float* __restrict__ Wh, const float* __restrict__ ssrc,
                               const float* __restrict__ sdst, const float* __restrict__ cn,
                               float* __restrict__ out_f, __bf16* __restrict__ out_bf,
                               int Nn, int d) {
  __shared__ float sm[256];
  __shared__ float alpha[256];
  int i = blockIdx.x, j = threadIdx.x;
  float e = ssrc[i] + sdst[j] + cn[i] + cn[j];
  e = e > 0.f ? e : 0.2f * e;                      // LeakyReLU(0.2)
  float m = block_reduce_max(e, sm);
  float p = __expf(e - m);
  float s = block_reduce_sum(p, sm);
  alpha[j] = p / s;
  __syncthreads();
  float acc = 0.f;
  for (int jj = 0; jj < Nn; ++jj)                  // coalesced rows of Wh (L2-hot)
    acc += alpha[jj] * Wh[(size_t)jj * d + j];
  acc = acc > 0.f ? acc : __expf(acc) - 1.f;       // ELU
  if (out_f)  out_f[(size_t)i * d + j] = acc;
  if (out_bf) out_bf[(size_t)i * d + j] = (__bf16)acc;
}

// ---------------------------------------------------------------------------
// Per-agent 2-layer head: out = W2 @ relu(W1 @ x_n + b1) + b2
// One block (8 wave32) per agent; 268MB of unique weights streamed float4-wide
// with shuffle reductions -> HBM-bandwidth bound (~23.3 TB/s roofline).
// ---------------------------------------------------------------------------
__global__ __launch_bounds__(256)
void agent_head_kernel(const float* __restrict__ x, const float* __restrict__ W1,
                       const float* __restrict__ b1, const float* __restrict__ W2,
                       const float* __restrict__ b2, float* __restrict__ out,
                       int K, int H1, int O) {
  __shared__ float xs[512];
  __shared__ float hs[256];
  int n = blockIdx.x, tid = threadIdx.x;
  for (int i = tid; i < K; i += blockDim.x) xs[i] = x[(size_t)n * K + i];
  __syncthreads();

  int wave = tid >> 5, lane = tid & 31, nw = blockDim.x >> 5;
  const float* W1n = W1 + (size_t)n * H1 * K;
  for (int h = wave; h < H1; h += nw) {
    const float4* wr = (const float4*)(W1n + (size_t)h * K);
    if (h + nw < H1) __builtin_prefetch(W1n + (size_t)(h + nw) * K, 0, 1);
    float acc = 0.f;
    for (int c = lane; c < (K >> 2); c += 32) {
      float4 w4 = wr[c];
      float4 x4 = ((const float4*)xs)[c];
      acc += w4.x * x4.x + w4.y * x4.y + w4.z * x4.z + w4.w * x4.w;
    }
    for (int off = 16; off > 0; off >>= 1) acc += __shfl_xor(acc, off, 32);
    if (lane == 0) {
      float v = acc + b1[(size_t)n * H1 + h];
      hs[h] = v > 0.f ? v : 0.f;
    }
  }
  __syncthreads();

  const float* W2n = W2 + (size_t)n * O * H1;
  for (int o = wave; o < O; o += nw) {
    float acc = 0.f;
    for (int h = lane; h < H1; h += 32) acc += hs[h] * W2n[(size_t)o * H1 + h];
    for (int off = 16; off > 0; off >>= 1) acc += __shfl_xor(acc, off, 32);
    if (lane == 0) out[(size_t)n * O + o] = acc + b2[(size_t)n * O + o];
  }
}

// ---------------------------------------------------------------------------
extern "C" void kernel_launch(void* const* d_in, const int* in_sizes, int n_in,
                              void* d_out, int out_size, void* d_ws, size_t ws_size,
                              hipStream_t stream) {
  constexpr int N = 256, OBS = 256, ACT = 16, H = 512, Z = 128, G = 256;
  constexpr int IN = OBS + G, HID = IN / 2;

  const float* obs   = (const float*)d_in[0];
  const float* hprev = (const float*)d_in[1];
  const float* eps   = (const float*)d_in[2];
  const float* Wp1 = (const float*)d_in[3];  const float* bp1 = (const float*)d_in[4];
  const float* Wp2 = (const float*)d_in[5];  const float* bp2 = (const float*)d_in[6];
  const float* We1 = (const float*)d_in[7];  const float* be1 = (const float*)d_in[8];
  const float* We2 = (const float*)d_in[9];  const float* be2 = (const float*)d_in[10];
  const float* Wd1 = (const float*)d_in[11]; const float* bd1 = (const float*)d_in[12];
  const float* Wd2 = (const float*)d_in[13]; const float* bd2 = (const float*)d_in[14];
  const float* Wih = (const float*)d_in[15]; const float* Whh = (const float*)d_in[16];
  const float* bih = (const float*)d_in[17]; const float* bhh = (const float*)d_in[18];
  const float* Wg1 = (const float*)d_in[19]; const float* as1 = (const float*)d_in[20];
  const float* ad1 = (const float*)d_in[21];
  const float* Wg2 = (const float*)d_in[22]; const float* as2 = (const float*)d_in[23];
  const float* ad2 = (const float*)d_in[24];
  const float* cw  = (const float*)d_in[25]; const float* cb  = (const float*)d_in[26];
  const float* Wa1 = (const float*)d_in[27]; const float* ba1 = (const float*)d_in[28];
  const float* Wa2 = (const float*)d_in[29]; const float* ba2 = (const float*)d_in[30];
  const float* Wc1 = (const float*)d_in[31]; const float* bc1 = (const float*)d_in[32];
  const float* Wc2 = (const float*)d_in[33]; const float* bc2 = (const float*)d_in[34];

  // d_out layout (reference tuple order)
  float* out      = (float*)d_out;
  float* o_hnew   = out;                       // N*H
  float* o_nll    = o_hnew + N * H;            // N
  float* o_kl     = o_nll + N;                 // N
  float* o_logits = o_kl + N;                  // N*ACT
  float* o_values = o_logits + N * ACT;        // N
  float* o_muq    = o_values + N;              // N*Z
  float* o_lvq    = o_muq + N * Z;             // N*Z
  float* o_z      = o_lvq + N * Z;             // N*Z
  float* o_vgat   = o_z + N * Z;               // N*G

  // workspace bump allocator (~14 MB)
  float* ws = (float*)d_ws;
  size_t off = 0;
  auto allocf = [&](size_t nf) { float* p = ws + off; off += nf; return p; };
  auto allocb = [&](size_t nb) { __bf16* p = (__bf16*)(ws + off); off += (nb + 1) / 2; return p; };

  // fp32 intermediates (consumed by non-GEMM kernels)
  float* pout  = allocf((size_t)N * 2 * Z);
  float* eout  = allocf((size_t)N * 2 * Z);
  float* dxout = allocf((size_t)N * 2 * OBS);
  float* gi    = allocf((size_t)N * 3 * H);
  float* ghb   = allocf((size_t)N * 3 * H);
  float* S1    = allocf(Z);
  float* S2    = allocf(Z);
  float* scal  = allocf(2);
  float* cnb   = allocf(N);
  float* Wh1   = allocf((size_t)N * G);
  float* ss1   = allocf(N); float* sd1 = allocf(N);
  float* Wh2   = allocf((size_t)N * G);
  float* ss2   = allocf(N); float* sd2 = allocf(N);
  float* acin  = allocf((size_t)N * IN);

  // bf16 GEMM operands
  __bf16* hprev_b = allocb((size_t)N * H);
  __bf16* xh_b    = allocb((size_t)N * (OBS + H));
  __bf16* ph_b    = allocb((size_t)N * H);
  __bf16* eh_b    = allocb((size_t)N * H);
  __bf16* zh_b    = allocb((size_t)N * (Z + H));
  __bf16* dh_b    = allocb((size_t)N * H);
  __bf16* oz_b    = allocb((size_t)N * (OBS + Z));
  __bf16* hnew_b  = allocb((size_t)N * H);
  __bf16* H1_b    = allocb((size_t)N * G);
  __bf16* Wp1b = allocb((size_t)H * H);
  __bf16* Wp2b = allocb((size_t)2 * Z * H);
  __bf16* We1b = allocb((size_t)H * (OBS + H));
  __bf16* We2b = allocb((size_t)2 * Z * H);
  __bf16* Wd1b = allocb((size_t)H * (Z + H));
  __bf16* Wd2b = allocb((size_t)2 * OBS * H);
  __bf16* Wihb = allocb((size_t)3 * H * (OBS + Z));
  __bf16* Whhb = allocb((size_t)3 * H * H);
  __bf16* Wg1b = allocb((size_t)G * H);
  __bf16* Wg2b = allocb((size_t)G * G);
  (void)ws_size; (void)n_in; (void)in_sizes; (void)out_size;

  auto cvt = [&](const float* src, __bf16* dst, size_t n) {
    int n4 = (int)(n / 4);
    cvt4_bf16_kernel<<<(n4 + 255) / 256, 256, 0, stream>>>(src, dst, n4);
  };
  auto gemm = [&](const __bf16* X, int K, const __bf16* W, const float* bias,
                  float* Of, __bf16* Ob, int Nn, int act) {
    dim3 grid(Nn / 128, N / 16);   // 4 waves/block, 2 N-tiles/wave
    gemm_bf16_wmma<<<grid, dim3(128), 0, stream>>>(X, W, bias, Of, Ob, Nn, K, act);
  };
  auto concat2b = [&](__bf16* dst, const float* a, int wa, const float* b, int wb) {
    int total = N * (wa + wb);
    concat2_bf16_kernel<<<(total + 255) / 256, 256, 0, stream>>>(dst, a, wa, b, wb, total);
  };

  // ---- one-time bf16 conversion of GEMM weights + h_prev ----
  cvt(hprev, hprev_b, (size_t)N * H);
  cvt(Wp1, Wp1b, (size_t)H * H);
  cvt(Wp2, Wp2b, (size_t)2 * Z * H);
  cvt(We1, We1b, (size_t)H * (OBS + H));
  cvt(We2, We2b, (size_t)2 * Z * H);
  cvt(Wd1, Wd1b, (size_t)H * (Z + H));
  cvt(Wd2, Wd2b, (size_t)2 * OBS * H);
  cvt(Wih, Wihb, (size_t)3 * H * (OBS + Z));
  cvt(Whh, Whhb, (size_t)3 * H * H);
  cvt(Wg1, Wg1b, (size_t)G * H);
  cvt(Wg2, Wg2b, (size_t)G * G);

  // ---- VRNN ----
  concat2b(xh_b, obs, OBS, hprev, H);                      // [obs | h_prev]
  gemm(hprev_b, H, Wp1b, bp1, nullptr, ph_b, H, 1);        // prior hidden (relu, bf16-only)
  gemm(ph_b, H, Wp2b, bp2, pout, nullptr, 2 * Z, 0);       // prior mu/lv
  gemm(xh_b, OBS + H, We1b, be1, nullptr, eh_b, H, 1);     // encoder hidden
  gemm(eh_b, H, We2b, be2, eout, nullptr, 2 * Z, 0);       // encoder mu/lv
  vrnnz_kernel<<<(N * Z + 255) / 256, 256, 0, stream>>>(eout, eps, o_muq, o_lvq, o_z, Z, N * Z);
  concat2b(zh_b, o_z, Z, hprev, H);                        // [z | h_prev]
  gemm(zh_b, Z + H, Wd1b, bd1, nullptr, dh_b, H, 1);       // decoder hidden
  gemm(dh_b, H, Wd2b, bd2, dxout, nullptr, 2 * OBS, 0);    // decoder mu/lv
  nll_kernel<<<N, 256, 0, stream>>>(obs, dxout, o_nll, OBS);
  kl_kernel<<<N, 128, 0, stream>>>(pout, o_muq, o_lvq, o_kl, Z);

  // ---- GRU ----
  concat2b(oz_b, obs, OBS, o_z, Z);                        // [obs | z]
  gemm(oz_b, OBS + Z, Wihb, bih, gi, nullptr, 3 * H, 0);
  gemm(hprev_b, H, Whhb, bhh, ghb, nullptr, 3 * H, 0);
  gru_kernel<<<(N * H + 255) / 256, 256, 0, stream>>>(gi, ghb, hprev, o_hnew, hnew_b, H, N * H);

  // ---- closed-form pairwise-KL coop + cn ----
  klstats_kernel<<<1, 128, 0, stream>>>(o_muq, o_lvq, S1, S2, scal, N, Z);
  coop_kernel<<<N, 128, 0, stream>>>(o_muq, o_lvq, S1, S2, scal, cw, cb, cnb, N, Z);

  // ---- 2-stage GAT ----
  gemm(hnew_b, H, Wg1b, nullptr, Wh1, nullptr, G, 0);
  gat_scores_kernel<<<N, 256, 0, stream>>>(Wh1, as1, ad1, ss1, sd1, G);
  gat_agg_kernel<<<N, 256, 0, stream>>>(Wh1, ss1, sd1, cnb, nullptr, H1_b, N, G);
  gemm(H1_b, G, Wg2b, nullptr, Wh2, nullptr, G, 0);
  gat_scores_kernel<<<N, 256, 0, stream>>>(Wh2, as2, ad2, ss2, sd2, G);
  gat_agg_kernel<<<N, 256, 0, stream>>>(Wh2, ss2, sd2, cnb, o_vgat, nullptr, N, G);

  // ---- per-agent actor/critic heads (HBM streaming) ----
  concat2_kernel<<<(N * IN + 255) / 256, 256, 0, stream>>>(acin, obs, OBS, o_vgat, G, N * IN);
  agent_head_kernel<<<N, 256, 0, stream>>>(acin, Wa1, ba1, Wa2, ba2, o_logits, IN, HID, ACT);
  agent_head_kernel<<<N, 256, 0, stream>>>(acin, Wc1, bc1, Wc2, bc2, o_values, IN, HID, 1);
}
